// CenterNet_38963943309824
// MI455X (gfx1250) — matrix operand order repaired and checked by reading, not compiled, not run on hardware
//
#include <hip/hip_runtime.h>
#include <hip/hip_bf16.h>
#include <math.h>

typedef __attribute__((ext_vector_type(16))) _Float16 v16h;
typedef __attribute__((ext_vector_type(8)))  float    v8f;

#define HH 128
#define WW 128
#define BB 4
#define NCLS 80
#define TW 66   // padded LDS tile width (64 + 2 halo)

// ---------------------------------------------------------------------------
// Workspace layout (bytes, all 256-aligned)
// ---------------------------------------------------------------------------
static const size_t OFF_ACT_IN = 0;            // f16 NHWC  4*128*128*128*2 = 16 MB
static const size_t OFF_ACT_A  = 16777216;
static const size_t OFF_ACT_B  = 33554432;
static const size_t OFF_LOGITS = 50331648;     // f32 NCHW 4*80*128*128*4 (reused as heatmap p)
static const size_t OFF_NMS    = 71303168;
static const size_t OFF_WH     = 92274688;
static const size_t OFF_OFF    = 92798976;
static const size_t OFF_PW0    = 93323264;     // packed weights, 9 x 294912 B (8 tiles each)
static const size_t OFF_PW1    = 93618176;
static const size_t OFF_PW2    = 93913088;
static const size_t OFF_PW3    = 94208000;
static const size_t OFF_PW4    = 94502912;
static const size_t OFF_PW5    = 94797824;
static const size_t OFF_PW6    = 95092736;
static const size_t OFF_PW7    = 95387648;
static const size_t OFF_PW8    = 95682560;

// ---------------------------------------------------------------------------
// f32 NCHW -> f16 NHWC conversion of the input feature map
// ---------------------------------------------------------------------------
__global__ __launch_bounds__(256) void cvt_in_kernel(const float* __restrict__ src,
                                                     _Float16* __restrict__ dst) {
  size_t idx = (size_t)blockIdx.x * 256 + threadIdx.x;   // B*H*W*128 = 8388608
  int c = idx & 127;
  size_t rest = idx >> 7;
  int w = rest & 127;
  int h = (rest >> 7) & 127;
  int b = rest >> 14;
  dst[idx] = (_Float16)src[((((size_t)b * 128 + c) * HH) + h) * WW + w];
}

// ---------------------------------------------------------------------------
// Pack OIHW f32 weights into WMMA A-fragment layout:
//   [tile16][kpos(9)][kchunk(4)][lane(32)][j(16)]  (f16)
// A k order used here (contiguous, not AMD's table): lane<16: K=j
//                     lane>=16: K=j+16;  M = lane&15
// ---------------------------------------------------------------------------
__global__ __launch_bounds__(256) void pack_w_kernel(const float* __restrict__ w,
                                                     _Float16* __restrict__ out,
                                                     int C_out, int total) {
  int idx = blockIdx.x * 256 + threadIdx.x;
  if (idx >= total) return;
  int j    = idx & 15;
  int lane = (idx >> 4) & 31;
  int kc   = (idx >> 9) & 3;
  int t    = idx >> 11;        // tile*9 + kpos
  int kpos = t % 9;
  int tile = t / 9;
  int m = lane & 15;
  int kk = (lane >> 4) * 16 + j;
  int ci = kc * 32 + kk;
  int co = tile * 16 + m;
  float v = 0.f;
  if (co < C_out)
    v = w[(((size_t)co * 128) + ci) * 9 + kpos];
  out[idx] = (_Float16)v;
}

// ---------------------------------------------------------------------------
// Implicit-GEMM 3x3 conv via v_wmma_f32_16x16x32_f16.
// Input: NHWC f16, C_in = 128. Output: NHWC f16 (trunk) or NCHW f32 (heads).
// Block tile: 128 C_out x 64 W at fixed (b, h). 8 waves, 2x2 register tile
// per wave -> 144 WMMAs/wave. LDS staged via GLOBAL_LOAD_ASYNC_TO_LDS_B128.
// ---------------------------------------------------------------------------
__global__ __launch_bounds__(256) void conv3x3_wmma_kernel(
    const _Float16* __restrict__ in, const _Float16* __restrict__ wpack,
    const float* __restrict__ bias, _Float16* __restrict__ out16,
    float* __restrict__ out32, int C_out, int relu) {
  __shared__ __align__(32) _Float16 tile[3 * TW * 128];   // 50688 B

  const int tid = threadIdx.x;
  int gx = blockIdx.x;              // (W/64)*H*B = 2*128*4 = 1024
  int wt = gx & 1;
  int rest = gx >> 1;
  int h = rest & 127;
  int b = rest >> 7;
  int w0 = wt * 64;

  // Stage input tile (3 rows x 66 cols x 128 ch) into LDS.
  // In-bounds 16B chunks: async copy global->LDS (ASYNCcnt). Halo: zero ds_store.
  for (int idx = tid; idx < 3 * TW * 16; idx += 256) {
    int pos = idx >> 4, ch = idx & 15;
    int r = pos / TW, c = pos % TW;
    int hh = h + r - 1, ww = w0 + c - 1;
    _Float16* lp = &tile[(r * TW + c) * 128 + ch * 8];
    if (hh >= 0 && hh < HH && ww >= 0 && ww < WW) {
      const _Float16* gp = in + ((((size_t)b * HH + hh) * WW + ww) << 7) + ch * 8;
      unsigned lds_addr = (unsigned)(size_t)lp;
      asm volatile("global_load_async_to_lds_b128 %0, %1, off"
                   :: "v"(lds_addr), "v"(gp) : "memory");
    } else {
      *reinterpret_cast<uint4*>(lp) = make_uint4(0u, 0u, 0u, 0u);
    }
  }
  asm volatile("s_wait_asynccnt 0" ::: "memory");
  __syncthreads();

  const int lane = tid & 31;
  const int wid = tid >> 5;
  const int wave_m = wid & 3;     // 4 waves x 32 C_out rows
  const int wave_n = wid >> 2;    // 2 waves x 32 W cols

  v8f acc[2][2] = {};
  const _Float16* wp0 = wpack + (size_t)(wave_m * 2)     * 36 * 512 + lane * 16;
  const _Float16* wp1 = wpack + (size_t)(wave_m * 2 + 1) * 36 * 512 + lane * 16;

  for (int kh = 0; kh < 3; ++kh) {
    for (int kw = 0; kw < 3; ++kw) {
      const _Float16* bp =
          &tile[((kh * TW) + wave_n * 32 + (lane & 15) + kw) * 128 + (lane >> 4) * 16];
      const int koff = (kh * 3 + kw) * 4 * 512;
#pragma unroll
      for (int kc = 0; kc < 4; ++kc) {
        v16h a0 = *reinterpret_cast<const v16h*>(wp0 + koff + kc * 512);
        v16h a1 = *reinterpret_cast<const v16h*>(wp1 + koff + kc * 512);
        v16h b0 = *reinterpret_cast<const v16h*>(bp + kc * 32);
        v16h b1 = *reinterpret_cast<const v16h*>(bp + 16 * 128 + kc * 32);
        acc[0][0] = __builtin_amdgcn_wmma_f32_16x16x32_f16(false, a0, false, b0, (short)0, acc[0][0], false, false);
        acc[0][1] = __builtin_amdgcn_wmma_f32_16x16x32_f16(false, a0, false, b1, (short)0, acc[0][1], false, false);
        acc[1][0] = __builtin_amdgcn_wmma_f32_16x16x32_f16(false, a1, false, b0, (short)0, acc[1][0], false, false);
        acc[1][1] = __builtin_amdgcn_wmma_f32_16x16x32_f16(false, a1, false, b1, (short)0, acc[1][1], false, false);
      }
    }
  }

  // Epilogue: D layout -> lane l holds rows m = r + (l>=16?8:0), col n = l&15
  const int mhi = (lane >> 4) * 8;
  const int nn = lane & 15;
#pragma unroll
  for (int m2 = 0; m2 < 2; ++m2) {
    const int co = (wave_m * 2 + m2) * 16 + mhi;
    if (co >= C_out) continue;
#pragma unroll
    for (int n2 = 0; n2 < 2; ++n2) {
      const int w = w0 + wave_n * 32 + n2 * 16 + nn;
      if (out16) {   // trunk: C_out == 128, NHWC f16
        __align__(16) _Float16 o[8];
#pragma unroll
        for (int r = 0; r < 8; ++r) {
          float v = acc[m2][n2][r] + bias[co + r];
          if (relu) v = v > 0.f ? v : 0.f;
          o[r] = (_Float16)v;
        }
        *reinterpret_cast<uint4*>(out16 + ((((size_t)b * HH + h) * WW + w) << 7) + co) =
            *reinterpret_cast<const uint4*>(o);
      } else {       // head: NCHW f32
#pragma unroll
        for (int r = 0; r < 8; ++r) {
          int c2 = co + r;
          if (c2 < C_out) {
            float v = acc[m2][n2][r] + bias[c2];
            if (relu) v = v > 0.f ? v : 0.f;
            out32[(((size_t)b * C_out + c2) * HH + h) * WW + w] = v;
          }
        }
      }
    }
  }
}

// ---------------------------------------------------------------------------
// Fused sigmoid + focal loss; overwrites logits buffer with p (heatmap).
// ---------------------------------------------------------------------------
__global__ __launch_bounds__(256) void focal_kernel(float* __restrict__ logits,
                                                    const float* __restrict__ targets,
                                                    float* __restrict__ loss) {
  __shared__ float red[256];
  size_t idx = (size_t)blockIdx.x * 256 + threadIdx.x;  // N = 5242880, exact
  float x = logits[idx];
  float t = targets[idx];
  float p = 1.f / (1.f + __expf(-x));
  float l1p = log1pf(__expf(-fabsf(x)));
  float logp   = fminf(x, 0.f) - l1p;    // log sigmoid(x)
  float log1mp = fminf(-x, 0.f) - l1p;   // log sigmoid(-x)
  float contrib = 0.f;
  if (t == 1.f) { float q = 1.f - p; contrib += logp * q * q; }
  if (t < 1.f)  { float nw = 1.f - t; nw *= nw; nw *= nw; contrib += log1mp * p * p * nw; }
  logits[idx] = p;
  red[threadIdx.x] = contrib;
  __syncthreads();
  for (int s = 128; s > 0; s >>= 1) {
    if (threadIdx.x < s) red[threadIdx.x] += red[threadIdx.x + s];
    __syncthreads();
  }
  if (threadIdx.x == 0) atomicAdd(loss, -red[0]);
}

// ---------------------------------------------------------------------------
// 3x3 max-pool NMS (SAME, -inf pad): out = (local max == v) ? v : 0
// ---------------------------------------------------------------------------
__global__ __launch_bounds__(256) void nms_kernel(const float* __restrict__ hm,
                                                  float* __restrict__ out) {
  size_t idx = (size_t)blockIdx.x * 256 + threadIdx.x;
  int x = idx & 127;
  int y = (idx >> 7) & 127;
  size_t plane = idx >> 14;   // b*80 + c
  float v = hm[idx];
  float mx = -INFINITY;
  for (int dy = -1; dy <= 1; ++dy)
    for (int dx = -1; dx <= 1; ++dx) {
      int yy = y + dy, xx = x + dx;
      if (yy >= 0 && yy < 128 && xx >= 0 && xx < 128)
        mx = fmaxf(mx, hm[(plane << 14) + yy * 128 + xx]);
    }
  out[idx] = (mx == v) ? v : 0.f;
}

// ---------------------------------------------------------------------------
// Per-batch top-100 (sequential argmax, lower-index tie break) + box decode.
// ---------------------------------------------------------------------------
__global__ __launch_bounds__(1024) void topk_kernel(float* __restrict__ hm_nms,
                                                    const float* __restrict__ wh,
                                                    const float* __restrict__ off,
                                                    float* __restrict__ out) {
  const int b = blockIdx.x;
  const int N = NCLS * HH * WW;   // 1310720
  float* data = hm_nms + (size_t)b * N;
  __shared__ float sv[1024];
  __shared__ int si[1024];
  const int tid = threadIdx.x;
  for (int k = 0; k < 100; ++k) {
    float best = -INFINITY;
    int bi = N;
    for (int i = tid; i < N; i += 1024) {
      float v = data[i];
      if (v > best || (v == best && i < bi)) { best = v; bi = i; }
    }
    sv[tid] = best; si[tid] = bi;
    __syncthreads();
    for (int s = 512; s > 0; s >>= 1) {
      if (tid < s) {
        if (sv[tid + s] > sv[tid] || (sv[tid + s] == sv[tid] && si[tid + s] < si[tid])) {
          sv[tid] = sv[tid + s]; si[tid] = si[tid + s];
        }
      }
      __syncthreads();
    }
    if (tid == 0) {
      int ind = si[0];
      float score = sv[0];
      data[ind] = -INFINITY;
      const int hw = HH * WW;
      int cls = ind / hw;
      int sp = ind % hw;
      float ys = (float)(sp / WW);
      float xs = (float)(sp % WW);
      float ow = wh[((size_t)b * 2) * hw + sp];
      float oh = wh[((size_t)b * 2 + 1) * hw + sp];
      float ox = off[((size_t)b * 2) * hw + sp];
      float oy = off[((size_t)b * 2 + 1) * hw + sp];
      float cx = xs + ox, cy = ys + oy;
      float* bx = out + ((size_t)b * 100 + k) * 4;
      bx[0] = cx - ow * 0.5f;
      bx[1] = cy - oh * 0.5f;
      bx[2] = cx + ow * 0.5f;
      bx[3] = cy + oh * 0.5f;
      out[1600 + b * 100 + k] = score;
      out[2000 + b * 100 + k] = (float)cls;
    }
    __syncthreads();
  }
}

__global__ void init_loss_kernel(float* out) {
  if (threadIdx.x == 0) out[2400] = 0.f;
}

// ---------------------------------------------------------------------------
extern "C" void kernel_launch(void* const* d_in, const int* in_sizes, int n_in,
                              void* d_out, int out_size, void* d_ws, size_t ws_size,
                              hipStream_t stream) {
  const float* feature      = (const float*)d_in[0];
  const float* targets      = (const float*)d_in[1];
  const float* cls_w        = (const float*)d_in[2];
  const float* cls_b        = (const float*)d_in[3];
  const float* bbox_w       = (const float*)d_in[4];
  const float* bbox_b       = (const float*)d_in[5];
  const float* off_w        = (const float*)d_in[6];
  const float* off_b        = (const float*)d_in[7];
  const float* cls_score_w  = (const float*)d_in[8];
  const float* cls_score_b  = (const float*)d_in[9];
  const float* bbox_pred_w  = (const float*)d_in[10];
  const float* bbox_pred_b  = (const float*)d_in[11];
  const float* center_off_w = (const float*)d_in[12];
  const float* center_off_b = (const float*)d_in[13];
  float* out = (float*)d_out;
  char* ws = (char*)d_ws;

  _Float16* actIn = (_Float16*)(ws + OFF_ACT_IN);
  _Float16* actA  = (_Float16*)(ws + OFF_ACT_A);
  _Float16* actB  = (_Float16*)(ws + OFF_ACT_B);
  float* logits   = (float*)(ws + OFF_LOGITS);
  float* hm_nms   = (float*)(ws + OFF_NMS);
  float* whb      = (float*)(ws + OFF_WH);
  float* offb     = (float*)(ws + OFF_OFF);
  _Float16* pw[9] = {
      (_Float16*)(ws + OFF_PW0), (_Float16*)(ws + OFF_PW1),
      (_Float16*)(ws + OFF_PW2), (_Float16*)(ws + OFF_PW3),
      (_Float16*)(ws + OFF_PW4), (_Float16*)(ws + OFF_PW5),
      (_Float16*)(ws + OFF_PW6), (_Float16*)(ws + OFF_PW7),
      (_Float16*)(ws + OFF_PW8)};

  init_loss_kernel<<<1, 32, 0, stream>>>(out);

  // Convert input feature to f16 NHWC
  cvt_in_kernel<<<32768, 256, 0, stream>>>(feature, actIn);

  // Pack weights into WMMA A-fragment layout (all padded to 8x16-row tiles)
  const size_t WCONV = 128 * 128 * 9;
  const int T8 = 8 * 9 * 4 * 512;
  pack_w_kernel<<<(T8 + 255) / 256, 256, 0, stream>>>(cls_w,          pw[0], 128, T8);
  pack_w_kernel<<<(T8 + 255) / 256, 256, 0, stream>>>(cls_w + WCONV,  pw[1], 128, T8);
  pack_w_kernel<<<(T8 + 255) / 256, 256, 0, stream>>>(bbox_w,         pw[2], 128, T8);
  pack_w_kernel<<<(T8 + 255) / 256, 256, 0, stream>>>(bbox_w + WCONV, pw[3], 128, T8);
  pack_w_kernel<<<(T8 + 255) / 256, 256, 0, stream>>>(off_w,          pw[4], 128, T8);
  pack_w_kernel<<<(T8 + 255) / 256, 256, 0, stream>>>(off_w + WCONV,  pw[5], 128, T8);
  pack_w_kernel<<<(T8 + 255) / 256, 256, 0, stream>>>(cls_score_w,    pw[6], NCLS, T8);
  pack_w_kernel<<<(T8 + 255) / 256, 256, 0, stream>>>(bbox_pred_w,    pw[7], 2, T8);
  pack_w_kernel<<<(T8 + 255) / 256, 256, 0, stream>>>(center_off_w,   pw[8], 2, T8);

  const dim3 cgrid(1024);   // (W/64)*H*B

  // cls branch
  conv3x3_wmma_kernel<<<cgrid, 256, 0, stream>>>(actIn, pw[0], cls_b,       actA, nullptr, 128, 1);
  conv3x3_wmma_kernel<<<cgrid, 256, 0, stream>>>(actA,  pw[1], cls_b + 128, actB, nullptr, 128, 1);
  conv3x3_wmma_kernel<<<cgrid, 256, 0, stream>>>(actB,  pw[6], cls_score_b, nullptr, logits, NCLS, 0);
  // bbox branch
  conv3x3_wmma_kernel<<<cgrid, 256, 0, stream>>>(actIn, pw[2], bbox_b,       actA, nullptr, 128, 1);
  conv3x3_wmma_kernel<<<cgrid, 256, 0, stream>>>(actA,  pw[3], bbox_b + 128, actB, nullptr, 128, 1);
  conv3x3_wmma_kernel<<<cgrid, 256, 0, stream>>>(actB,  pw[7], bbox_pred_b,  nullptr, whb, 2, 0);
  // offset branch
  conv3x3_wmma_kernel<<<cgrid, 256, 0, stream>>>(actIn, pw[4], off_b,        actA, nullptr, 128, 1);
  conv3x3_wmma_kernel<<<cgrid, 256, 0, stream>>>(actA,  pw[5], off_b + 128,  actB, nullptr, 128, 1);
  conv3x3_wmma_kernel<<<cgrid, 256, 0, stream>>>(actB,  pw[8], center_off_b, nullptr, offb, 2, 0);

  // Focal loss + sigmoid (logits -> p in place), NMS, top-k + decode
  focal_kernel<<<20480, 256, 0, stream>>>(logits, targets, out + 2400);
  nms_kernel<<<20480, 256, 0, stream>>>(logits, hm_nms);
  topk_kernel<<<BB, 1024, 0, stream>>>(hm_nms, whb, offb, out);
}
